// SelfAttentionBlock_35837207118438
// MI455X (gfx1250) — matrix operand order repaired and checked by reading, not compiled, not run on hardware
//
#include <hip/hip_runtime.h>
#include <hip/hip_bf16.h>

// ---------------------------------------------------------------------------
// SelfAttentionBlock for MI455X (gfx1250): GroupNorm -> QKV -> MHA -> proj
// All matmuls on v_wmma_f32_16x16x32_f16 (wave32 WMMA, f32 accum).
// Round 2: software-pipelined GEMM k-loops, 32x64 wave tiles, double-buffered
// K/V LDS staging in attention, global_prefetch for upcoming chunks.
// ---------------------------------------------------------------------------

typedef __attribute__((ext_vector_type(16))) _Float16 v16h;
typedef __attribute__((ext_vector_type(8)))  _Float16 v8h;
typedef __attribute__((ext_vector_type(8)))  float    v8f;

#define CC   256
#define HW   4096      // 64*64
#define NB   4         // batch
#define NH   4         // heads
#define HD   64        // head dim
#define SEQ  4096

// ---- WMMA fragment loaders (contiguous k = 16h + s, same in A and B) ------------------
// A 16x32 f16: lane row m = lane&15 ; lane holds k in [h16,h16+16),
// h16=(lane>>4)*16.  Row-major [m][k] -> two 16B loads.
__device__ __forceinline__ v16h load_a_frag(const _Float16* row_base, int k0, int lane) {
    int h8 = (lane >> 4) << 4;
    const _Float16* p = row_base + k0 + h8;
    v8h lo = *(const v8h*)(p);
    v8h hi = *(const v8h*)(p + 8);
    v16h a;
#pragma unroll
    for (int i = 0; i < 8; ++i) { a[i] = lo[i]; a[i + 8] = hi[i]; }
    return a;
}

// B 32x16 f16 with memory holding B^T row-major [n][k] (k contiguous):
// lane col n = lane&15 ; k in [koff,koff+16), koff=(lane>>4)*16 -> one 32B load.
__device__ __forceinline__ v16h load_bt_frag(const _Float16* base, int n0, int ld,
                                             int k0, int lane) {
    int n    = lane & 15;
    int koff = (lane >> 4) << 4;
    return *(const v16h*)(base + (size_t)(n0 + n) * ld + k0 + koff);
}

__device__ __forceinline__ v8f wmma_16x16x32(v16h a, v16h b, v8f c) {
    return __builtin_amdgcn_wmma_f32_16x16x32_f16(false, a, false, b,
                                                  (short)0, c, false, false);
}

// ---- 1) GroupNorm statistics ----------------------------------------------
__global__ void sab_gn_stats(const float* __restrict__ x, float* __restrict__ stats) {
    int bid = blockIdx.x;            // n*8 + g
    int n = bid >> 3, g = bid & 7;
    const float* base = x + ((size_t)n * CC + (size_t)g * 32) * HW;
    float s = 0.f, ss = 0.f;
    for (int i = threadIdx.x; i < 32 * HW; i += 256) {
        float v = base[i]; s += v; ss += v * v;
    }
    __shared__ float rs[256], rq[256];
    rs[threadIdx.x] = s; rq[threadIdx.x] = ss;
    __syncthreads();
    for (int o = 128; o > 0; o >>= 1) {
        if (threadIdx.x < o) { rs[threadIdx.x] += rs[threadIdx.x + o];
                               rq[threadIdx.x] += rq[threadIdx.x + o]; }
        __syncthreads();
    }
    if (threadIdx.x == 0) {
        const float inv = 1.0f / (32.0f * HW);
        float mean = rs[0] * inv;
        float var  = rq[0] * inv - mean * mean;
        stats[2 * bid]     = mean;
        stats[2 * bid + 1] = rsqrtf(var + 1e-5f);
    }
}

// ---- 2) normalize, scale/shift, transpose NCHW -> [n*p][C], cast f16 -------
__global__ void sab_gn_apply(const float* __restrict__ x,
                             const float* __restrict__ w,
                             const float* __restrict__ b,
                             const float* __restrict__ stats,
                             _Float16* __restrict__ xn) {
    int i = blockIdx.x * 256 + threadIdx.x;       // over N*C*HW
    int p = i & (HW - 1);
    int c = (i >> 12) & (CC - 1);
    int n = i >> 20;
    int sid = n * 8 + (c >> 5);
    float mean = stats[2 * sid], rstd = stats[2 * sid + 1];
    float v = (x[i] - mean) * rstd * w[c] + b[c];
    xn[((size_t)(n * HW + p)) * CC + c] = (_Float16)v;
}

// ---- 3) convert weights to f16 (row-major [o][c], K contiguous) ------------
__global__ void sab_wcvt(const float* __restrict__ qkvw, const float* __restrict__ projw,
                         _Float16* __restrict__ wq, _Float16* __restrict__ wp) {
    int i = blockIdx.x * 256 + threadIdx.x;
    if (i < 3 * CC * CC) wq[i] = (_Float16)qkvw[i];
    else                 wp[i - 3 * CC * CC] = (_Float16)projw[i - 3 * CC * CC];
}

// ---- 4) QKV GEMM: [16384 x 256] @ [256 x 768] + b, scatter to heads --------
// grid (M/256, 768/64), 8 waves; wave tile 32x64, software-pipelined k-loop.
__global__ __launch_bounds__(256) void sab_qkv_gemm(
        const _Float16* __restrict__ xn,
        const _Float16* __restrict__ wq,
        const float* __restrict__ bias,
        _Float16* __restrict__ q,
        _Float16* __restrict__ k,
        _Float16* __restrict__ v) {
    int wave = threadIdx.x >> 5, lane = threadIdx.x & 31;
    int m0 = blockIdx.x * 256 + wave * 32;
    int n0 = blockIdx.y * 64;
    v8f acc[2][4] = {};
    const _Float16* arow0 = xn + (size_t)(m0 + (lane & 15)) * CC;
    const _Float16* arow1 = arow0 + (size_t)16 * CC;

    v16h a0 = load_a_frag(arow0, 0, lane);
    v16h a1 = load_a_frag(arow1, 0, lane);
    v16h b[4];
#pragma unroll
    for (int nt = 0; nt < 4; ++nt) b[nt] = load_bt_frag(wq, n0 + nt * 16, CC, 0, lane);

#pragma unroll
    for (int k0 = 0; k0 < CC; k0 += 32) {
        v16h na0 = a0, na1 = a1, nb[4];
        if (k0 + 32 < CC) {
            na0 = load_a_frag(arow0, k0 + 32, lane);
            na1 = load_a_frag(arow1, k0 + 32, lane);
#pragma unroll
            for (int nt = 0; nt < 4; ++nt)
                nb[nt] = load_bt_frag(wq, n0 + nt * 16, CC, k0 + 32, lane);
        }
#pragma unroll
        for (int nt = 0; nt < 4; ++nt) {
            acc[0][nt] = wmma_16x16x32(a0, b[nt], acc[0][nt]);
            acc[1][nt] = wmma_16x16x32(a1, b[nt], acc[1][nt]);
        }
        if (k0 + 32 < CC) {
            a0 = na0; a1 = na1;
#pragma unroll
            for (int nt = 0; nt < 4; ++nt) b[nt] = nb[nt];
        }
    }

#pragma unroll
    for (int mt = 0; mt < 2; ++mt) {
        int mbase = m0 + mt * 16 + ((lane >> 4) << 3);
#pragma unroll
        for (int nt = 0; nt < 4; ++nt) {
            int col = n0 + nt * 16 + (lane & 15);
            float bv = bias[col];
            int which = col >> 8;            // 0=q 1=k 2=v
            int rem = col & 255;
            int head = rem >> 6, d = rem & 63;
            _Float16* dst = which == 0 ? q : (which == 1 ? k : v);
#pragma unroll
            for (int r = 0; r < 8; ++r) {
                int m  = mbase + r;
                int nb2 = m >> 12, p = m & (HW - 1);
                dst[((size_t)((nb2 * NH + head) * SEQ + p)) * HD + d] =
                    (_Float16)(acc[mt][nt][r] + bv);
            }
        }
    }
}

// ---- 5) Flash attention, one (batch,head) per blockIdx.y -------------------
// 8 waves x 16 q-rows = 128 q-rows per block; kv chunks of 64, double-buffered
// LDS staging + global prefetch two chunks ahead.
__global__ __launch_bounds__(256) void sab_attn(
        const _Float16* __restrict__ Q,
        const _Float16* __restrict__ K,
        const _Float16* __restrict__ V,
        _Float16* __restrict__ out) {
    const int bh   = blockIdx.y;                 // 0..15
    const int wave = threadIdx.x >> 5, lane = threadIdx.x & 31;
    const int t    = threadIdx.x;
    __shared__ __attribute__((aligned(32))) _Float16 Ks[2][64 * 80];   // [kv][dim]
    __shared__ __attribute__((aligned(32))) _Float16 Vt[2][64 * 80];   // [dim][kv]
    __shared__ __attribute__((aligned(32))) _Float16 Pb[8 * 16 * 72];  // per-wave P

    const size_t seqbase = (size_t)bh * SEQ * HD;
    const _Float16* Kg = K + seqbase;
    const _Float16* Vg = V + seqbase;
    const int qrow0 = blockIdx.x * 128 + wave * 16;

    const _Float16* qrow = Q + seqbase + (size_t)(qrow0 + (lane & 15)) * HD;
    v16h qa0 = load_a_frag(qrow, 0,  lane);
    v16h qa1 = load_a_frag(qrow, 32, lane);

    v8f acc[4] = {};
    float mrun[8], lrun[8];
#pragma unroll
    for (int r = 0; r < 8; ++r) { mrun[r] = -1e30f; lrun[r] = 0.f; }
    _Float16* Pw = Pb + wave * 16 * 72;
    const float scale = 0.125f;                  // 1/sqrt(64)

    const int rr = t >> 2;                        // staging row   0..63
    const int dg = (t & 3) << 4;                  // staging dim group

    // ---- stage chunk 0 ----
    {
        v16h kreg = *(const v16h*)(Kg + (size_t)rr * HD + dg);
        v16h vreg = *(const v16h*)(Vg + (size_t)rr * HD + dg);
        *(v16h*)(&Ks[0][rr * 80 + dg]) = kreg;
#pragma unroll
        for (int i = 0; i < 16; ++i) Vt[0][(dg + i) * 80 + rr] = vreg[i];
    }
    __syncthreads();

    const int NCHUNK = SEQ / 64;
    for (int ic = 0; ic < NCHUNK; ++ic) {
        const int cur = ic & 1;
        const bool more = (ic + 1) < NCHUNK;
        v16h kreg, vreg;
        if (more) {                               // global loads for next chunk
            size_t off = (size_t)((ic + 1) * 64 + rr) * HD + dg;
            kreg = *(const v16h*)(Kg + off);
            vreg = *(const v16h*)(Vg + off);
        }
        if (ic + 2 < NCHUNK) {                    // prefetch two chunks ahead
            size_t poff = (size_t)((ic + 2) * 64 + rr) * HD + dg;
            __builtin_prefetch(Kg + poff, 0, 1);
            __builtin_prefetch(Vg + poff, 0, 1);
        }
        const _Float16* KsC = Ks[cur];
        const _Float16* VtC = Vt[cur];

        // ---- S = Q K^T : 16 rows x 64 kv cols ----
        v8f s[4] = {};
#pragma unroll
        for (int nt = 0; nt < 4; ++nt) {
            s[nt] = wmma_16x16x32(qa0, load_bt_frag(KsC, nt * 16, 80, 0,  lane), s[nt]);
            s[nt] = wmma_16x16x32(qa1, load_bt_frag(KsC, nt * 16, 80, 32, lane), s[nt]);
        }

        // ---- online softmax (rows live across 16-lane halves) ----
#pragma unroll
        for (int r = 0; r < 8; ++r) {
            float rowm = -1e30f;
#pragma unroll
            for (int nt = 0; nt < 4; ++nt) {
                float val = s[nt][r] * scale; s[nt][r] = val;
                rowm = fmaxf(rowm, val);
            }
#pragma unroll
            for (int off = 8; off > 0; off >>= 1)
                rowm = fmaxf(rowm, __shfl_xor(rowm, off, 16));
            float newm = fmaxf(mrun[r], rowm);
            float corr = __expf(mrun[r] - newm);
            mrun[r] = newm;
            float psum = 0.f;
#pragma unroll
            for (int nt = 0; nt < 4; ++nt) {
                float pv = __expf(s[nt][r] - newm); s[nt][r] = pv; psum += pv;
            }
#pragma unroll
            for (int off = 8; off > 0; off >>= 1)
                psum += __shfl_xor(psum, off, 16);
            lrun[r] = lrun[r] * corr + psum;
#pragma unroll
            for (int nt = 0; nt < 4; ++nt) acc[nt][r] *= corr;
        }

        // ---- convert P (C/D layout) -> A layout via per-wave LDS ----
        {
            int rbase = (lane >> 4) << 3;
            int col   = lane & 15;
#pragma unroll
            for (int nt = 0; nt < 4; ++nt)
#pragma unroll
                for (int r = 0; r < 8; ++r)
                    Pw[(rbase + r) * 72 + nt * 16 + col] = (_Float16)s[nt][r];
        }

        // ---- O += P V ----
        const _Float16* prow = Pw + (size_t)(lane & 15) * 72;
#pragma unroll
        for (int ks = 0; ks < 2; ++ks) {
            v16h pa = load_a_frag(prow, ks * 32, lane);
#pragma unroll
            for (int nt = 0; nt < 4; ++nt) {
                v16h bv = load_bt_frag(VtC, nt * 16, 80, ks * 32, lane);
                acc[nt] = wmma_16x16x32(pa, bv, acc[nt]);
            }
        }

        // ---- stage next chunk into the other buffer ----
        if (more) {
            _Float16* Kn = Ks[cur ^ 1];
            _Float16* Vn = Vt[cur ^ 1];
            *(v16h*)(&Kn[rr * 80 + dg]) = kreg;
#pragma unroll
            for (int i = 0; i < 16; ++i) Vn[(dg + i) * 80 + rr] = vreg[i];
        }
        __syncthreads();
    }

    // ---- epilogue: normalize and scatter to [n][p][C] f16 ----
    int nb = bh >> 2, head = bh & 3;
    int rbase = (lane >> 4) << 3;
#pragma unroll
    for (int nt = 0; nt < 4; ++nt) {
        int d = nt * 16 + (lane & 15);
#pragma unroll
        for (int r = 0; r < 8; ++r) {
            int p = qrow0 + rbase + r;
            float val = acc[nt][r] / lrun[r];
            out[((size_t)(nb * HW + p)) * CC + head * HD + d] = (_Float16)val;
        }
    }
}

// ---- 6) proj GEMM + residual + mask (32x64 wave tile, pipelined) -----------
__global__ __launch_bounds__(256) void sab_proj_gemm(
        const _Float16* __restrict__ ao,
        const _Float16* __restrict__ wp,
        const float* __restrict__ bias,
        const float* __restrict__ x,
        const float* __restrict__ mask,
        float* __restrict__ y) {
    int wave = threadIdx.x >> 5, lane = threadIdx.x & 31;
    int m0 = blockIdx.x * 256 + wave * 32;
    int n0 = blockIdx.y * 64;
    v8f acc[2][4] = {};
    const _Float16* arow0 = ao + (size_t)(m0 + (lane & 15)) * CC;
    const _Float16* arow1 = arow0 + (size_t)16 * CC;

    v16h a0 = load_a_frag(arow0, 0, lane);
    v16h a1 = load_a_frag(arow1, 0, lane);
    v16h b[4];
#pragma unroll
    for (int nt = 0; nt < 4; ++nt) b[nt] = load_bt_frag(wp, n0 + nt * 16, CC, 0, lane);

#pragma unroll
    for (int k0 = 0; k0 < CC; k0 += 32) {
        v16h na0 = a0, na1 = a1, nb[4];
        if (k0 + 32 < CC) {
            na0 = load_a_frag(arow0, k0 + 32, lane);
            na1 = load_a_frag(arow1, k0 + 32, lane);
#pragma unroll
            for (int nt = 0; nt < 4; ++nt)
                nb[nt] = load_bt_frag(wp, n0 + nt * 16, CC, k0 + 32, lane);
        }
#pragma unroll
        for (int nt = 0; nt < 4; ++nt) {
            acc[0][nt] = wmma_16x16x32(a0, b[nt], acc[0][nt]);
            acc[1][nt] = wmma_16x16x32(a1, b[nt], acc[1][nt]);
        }
        if (k0 + 32 < CC) {
            a0 = na0; a1 = na1;
#pragma unroll
            for (int nt = 0; nt < 4; ++nt) b[nt] = nb[nt];
        }
    }

#pragma unroll
    for (int mt = 0; mt < 2; ++mt) {
        int mbase = m0 + mt * 16 + ((lane >> 4) << 3);
#pragma unroll
        for (int nt = 0; nt < 4; ++nt) {
            int c = n0 + nt * 16 + (lane & 15);
            float bv = bias[c];
#pragma unroll
            for (int r = 0; r < 8; ++r) {
                int m  = mbase + r;
                int nb2 = m >> 12, p = m & (HW - 1);
                size_t xi = ((size_t)nb2 * CC + c) * HW + p;
                y[xi] = (x[xi] + acc[mt][nt][r] + bv) * mask[nb2 * HW + p];
            }
        }
    }
}

// ---- 7) second tuple output: mask pass-through -----------------------------
__global__ void sab_copy_mask(const float* __restrict__ mask, float* __restrict__ dst) {
    int i = blockIdx.x * 256 + threadIdx.x;
    dst[i] = mask[i];
}

// ---------------------------------------------------------------------------
extern "C" void kernel_launch(void* const* d_in, const int* in_sizes, int n_in,
                              void* d_out, int out_size, void* d_ws, size_t ws_size,
                              hipStream_t stream) {
    const float* x     = (const float*)d_in[0];
    const float* mask  = (const float*)d_in[1];
    const float* nw    = (const float*)d_in[2];
    const float* nbi   = (const float*)d_in[3];
    const float* qkvw  = (const float*)d_in[4];
    const float* qkvb  = (const float*)d_in[5];
    const float* projw = (const float*)d_in[6];
    const float* projb = (const float*)d_in[7];

    char* ws = (char*)d_ws;
    constexpr size_t SZ16 = (size_t)NB * HW * CC * 2;   // 8 MB per f16 tensor
    float*    stats = (float*)(ws);
    _Float16* xn    = (_Float16*)(ws + 256);
    _Float16* q     = (_Float16*)(ws + 256 + 1 * SZ16);
    _Float16* k     = (_Float16*)(ws + 256 + 2 * SZ16);
    _Float16* v     = (_Float16*)(ws + 256 + 3 * SZ16);
    _Float16* ao    = (_Float16*)(ws + 256 + 4 * SZ16);
    _Float16* wq    = (_Float16*)(ws + 256 + 5 * SZ16);
    _Float16* wp    = (_Float16*)(ws + 256 + 5 * SZ16 + (size_t)3 * CC * CC * 2);

    sab_gn_stats <<<NB * 8, 256, 0, stream>>>(x, stats);
    sab_gn_apply <<<(NB * CC * HW) / 256, 256, 0, stream>>>(x, nw, nbi, stats, xn);
    sab_wcvt     <<<(4 * CC * CC) / 256, 256, 0, stream>>>(qkvw, projw, wq, wp);
    sab_qkv_gemm <<<dim3(64, 12), 256, 0, stream>>>(xn, wq, qkvb, q, k, v);
    sab_attn     <<<dim3(SEQ / 128, NB * NH), 256, 0, stream>>>(q, k, v, ao);
    sab_proj_gemm<<<dim3(64, 4), 256, 0, stream>>>(ao, wp, projb, x, mask, (float*)d_out);
    sab_copy_mask<<<(NB * HW) / 256, 256, 0, stream>>>(mask, (float*)d_out + (size_t)NB * CC * HW);
}
